// UnifiedSelfAttentionTransformer_65481071398677
// MI455X (gfx1250) — matrix operand order repaired and checked by reading, not compiled, not run on hardware
//
#include <hip/hip_runtime.h>
#include <hip/hip_bf16.h>
#include <math.h>

// ---- problem constants (match reference) ----
#define BQ   8      // batch
#define SEQ  1024   // sequence
#define EMB  256    // embedding E
#define NH   8      // heads
#define DKQ  32     // dk = E/H
#define NL   4      // layers
#define FF   1024   // ffn hidden
#define NQQ  256    // quantum wires == E
#define NC   10     // classes

typedef __bf16 bf16_t;
typedef __attribute__((ext_vector_type(16))) __bf16 v16bf;
typedef __attribute__((ext_vector_type(8)))  __bf16 v8bf;
typedef __attribute__((ext_vector_type(8)))  float  v8f;

// float -> bf16 (round-to-nearest-even), storage-only use of __bf16
static __device__ __forceinline__ bf16_t f2bf(float f) {
  union { float f; unsigned u; } v; v.f = f;
  unsigned r = v.u + 0x7FFFu + ((v.u >> 16) & 1u);
  unsigned short h = (unsigned short)(r >> 16);
  bf16_t b;
  __builtin_memcpy(&b, &h, 2);
  return b;
}

// ---------------- WMMA fragment loaders (CDNA5 ISA 7.12.2 layouts) ----------------
// A: 16x32 bf16, row-major, lane L<16: M=L, K = {0..7} U {16..23}; lane>=16: M=L-16, K={8..15} U {24..31}
static __device__ __forceinline__ v16bf load_frag_a(const bf16_t* __restrict__ A, int lda,
                                                    int row0, int k0, int lane) {
  int r    = lane & 15;
  int ksel = (lane >> 4) * 8;
  const bf16_t* p0 = A + (size_t)(row0 + r) * lda + k0 + ksel;
  v8bf lo = *(const v8bf*)(p0);        // K = ksel + 0..7
  v8bf hi = *(const v8bf*)(p0 + 16);   // K = ksel + 16..23
  v16bf a;
#pragma unroll
  for (int i = 0; i < 8; ++i) { a[i] = lo[i]; a[8 + i] = hi[i]; }
  return a;
}

// B: 32x16 bf16. Fed from BT (NxK row-major): lane L<16: N=L, K = {0..7} U {16..23}; lane>=16: N=L-16, K = {8..15} U {24..31}
static __device__ __forceinline__ v16bf load_frag_b(const bf16_t* __restrict__ BT, int ldbt,
                                                    int col0, int k0, int lane) {
  int n    = lane & 15;
  int ksel = (lane >> 4) * 8;
  const bf16_t* p0 = BT + (size_t)(col0 + n) * ldbt + k0 + ksel;
  v8bf lo = *(const v8bf*)(p0);
  v8bf hi = *(const v8bf*)(p0 + 16);
  v16bf b;
#pragma unroll
  for (int i = 0; i < 8; ++i) { b[i] = lo[i]; b[8 + i] = hi[i]; }
  return b;
}

static __device__ __forceinline__ v8f wmma_bf16(v16bf a, v16bf b, v8f c) {
  return __builtin_amdgcn_wmma_f32_16x16x32_bf16(false, a, false, b, (short)0, c,
                                                 false, false);
}

// epilogue for one 16x16 tile
template <int HAS_BIAS, int RELU, int OUT_F32, int OUT_BF16>
static __device__ __forceinline__ void store_tile(
    v8f acc, float* __restrict__ Cf, bf16_t* __restrict__ Cb,
    const float* __restrict__ bias, size_t cbase, int ldc, int row0, int col0, int lane) {
  // C layout: VGPR i, lanes 0-15 -> M=i, N=lane; lanes 16-31 -> M=8+i, N=lane-16
  const int n     = col0 + (lane & 15);
  const int rbase = row0 + ((lane >> 4) ? 8 : 0);
  float bv = 0.f;
  if (HAS_BIAS) bv = bias[n];
#pragma unroll
  for (int i = 0; i < 8; ++i) {
    int r = rbase + i;
    float v = acc[i] + bv;
    if (RELU) v = fmaxf(v, 0.f);
    if (OUT_F32)  Cf[cbase + (size_t)r * ldc + n] = v;
    if (OUT_BF16) Cb[cbase + (size_t)r * ldc + n] = f2bf(v);
  }
}

// ---------------- generic bf16 WMMA GEMM: C[M,N] = A[M,K] * BT[N,K]^T (+bias)(+relu) ----------------
// grid: (N/128, M/64, batch); block: 256 threads = 8 waves laid out 2(M) x 4(N).
// Each wave register-blocks a 32x32 tile: 2x2 accumulators, 4 WMMAs per K-step
// (1 fragment load per WMMA instead of 2 -> half the VMEM traffic per matrix op).
template <int HAS_BIAS, int RELU, int OUT_F32, int OUT_BF16>
__global__ __launch_bounds__(256) void gemm_bf16_wmma(
    const bf16_t* __restrict__ A, const bf16_t* __restrict__ BT,
    float* __restrict__ Cf, bf16_t* __restrict__ Cb,
    const float* __restrict__ bias,
    int M, int N, int K, int lda, int ldbt, int ldc,
    long strideA, long strideBT, long strideC)
{
  const int lane = threadIdx.x & 31;
  const int wave = threadIdx.x >> 5;                    // 0..7
  const int row0 = blockIdx.y * 64  + (wave & 1) * 32;  // M position of 32x32 wave tile
  const int col0 = blockIdx.x * 128 + (wave >> 1) * 32; // N position
  const int bz   = blockIdx.z;
  if (row0 >= M || col0 >= N) return;                   // wave-uniform -> EXEC stays all-ones

  const bf16_t* Ab = A  + (size_t)bz * strideA;
  const bf16_t* Bb = BT + (size_t)bz * strideBT;

  v8f acc00 = {}, acc01 = {}, acc10 = {}, acc11 = {};
  for (int k0 = 0; k0 < K; k0 += 32) {
    if (k0 + 32 < K) {                                  // global_prefetch_b8 of next K slab
      __builtin_prefetch(Ab + (size_t)(row0 + (lane & 15)) * lda  + k0 + 32, 0, 1);
      __builtin_prefetch(Bb + (size_t)(col0 + (lane & 15)) * ldbt + k0 + 32, 0, 1);
    }
    v16bf a0 = load_frag_a(Ab, lda,  row0,      k0, lane);
    v16bf a1 = load_frag_a(Ab, lda,  row0 + 16, k0, lane);
    v16bf b0 = load_frag_b(Bb, ldbt, col0,      k0, lane);
    v16bf b1 = load_frag_b(Bb, ldbt, col0 + 16, k0, lane);
    acc00 = wmma_bf16(a0, b0, acc00);
    acc01 = wmma_bf16(a0, b1, acc01);
    acc10 = wmma_bf16(a1, b0, acc10);
    acc11 = wmma_bf16(a1, b1, acc11);
  }

  const size_t cbase = (size_t)bz * strideC;
  store_tile<HAS_BIAS, RELU, OUT_F32, OUT_BF16>(acc00, Cf, Cb, bias, cbase, ldc, row0,      col0,      lane);
  store_tile<HAS_BIAS, RELU, OUT_F32, OUT_BF16>(acc01, Cf, Cb, bias, cbase, ldc, row0,      col0 + 16, lane);
  store_tile<HAS_BIAS, RELU, OUT_F32, OUT_BF16>(acc10, Cf, Cb, bias, cbase, ldc, row0 + 16, col0,      lane);
  store_tile<HAS_BIAS, RELU, OUT_F32, OUT_BF16>(acc11, Cf, Cb, bias, cbase, ldc, row0 + 16, col0 + 16, lane);
}

// ---------------- elementwise / reduction kernels ----------------
__global__ void embed_kernel(const int* __restrict__ tokens, const float* __restrict__ emb,
                             float* __restrict__ x) {
  int i = blockIdx.x * 256 + threadIdx.x;            // over B*S*E
  if (i >= BQ * SEQ * EMB) return;
  int e  = i % EMB;
  int bs = i / EMB;
  int s  = bs % SEQ;
  int tok = tokens[bs];
  int half = e >> 1;
  float div = __expf((float)(2 * half) * (-9.210340371976184f / (float)EMB)); // ln(10000)
  float arg = (float)s * div;
  float pe  = (e & 1) ? cosf(arg) : sinf(arg);
  x[i] = emb[(size_t)tok * EMB + e] + pe;
}

__global__ void transpose_to_bf16(const float* __restrict__ src, bf16_t* __restrict__ dst,
                                  int rows, int cols) {
  size_t mat  = (size_t)rows * cols;
  size_t base = (size_t)blockIdx.z * mat;
  int idx = blockIdx.x * 256 + threadIdx.x;
  if (idx >= rows * cols) return;
  int r = idx / cols, c = idx % cols;
  dst[base + (size_t)c * rows + r] = f2bf(src[base + idx]);
}

// quantum attention projection: per (b,s,h): c=cos(x+theta); out[0]=prod(c[1:]); out[i>=1]=cumprod(c)[i]
__global__ void qproj_kernel(const float* __restrict__ x, const float* __restrict__ theta,
                             bf16_t* __restrict__ pbf, bf16_t* __restrict__ pT) {
  int t = blockIdx.x * 256 + threadIdx.x;            // over B*S*H
  if (t >= BQ * SEQ * NH) return;
  int h  = t % NH;
  int bs = t / NH;
  int b  = bs / SEQ;
  int s  = bs % SEQ;
  const float* xin = x + (size_t)bs * EMB + h * DKQ;
  const float* th  = theta + h * DKQ;
  float c[DKQ], out[DKQ];
#pragma unroll
  for (int j = 0; j < DKQ; ++j) c[j] = cosf(xin[j] + th[j]);
  float tail = 1.f;
#pragma unroll
  for (int j = 1; j < DKQ; ++j) tail *= c[j];
  out[0] = tail;
  float cum = c[0];
#pragma unroll
  for (int j = 1; j < DKQ; ++j) { cum *= c[j]; out[j] = cum; }
  bf16_t* prow = pbf + (size_t)bs * EMB + h * DKQ;
  bf16_t* pTb  = pT + (size_t)b * EMB * SEQ;
#pragma unroll
  for (int j = 0; j < DKQ; ++j) {
    bf16_t v = f2bf(out[j]);
    prow[j] = v;
    pTb[(size_t)(h * DKQ + j) * SEQ + s] = v;        // transposed copy for the value GEMM
  }
}

// row softmax over S=1024 with scale applied pre-exp; fp32 in, bf16 out
__global__ void softmax_kernel(const float* __restrict__ Sf, bf16_t* __restrict__ Sb, float scale) {
  __shared__ float red[256];
  size_t row = blockIdx.x;                           // B*S rows
  const float* src = Sf + row * SEQ;
  bf16_t* dst = Sb + row * SEQ;
  int t = threadIdx.x;
  float v[4];
  float mx = -3.4e38f;
#pragma unroll
  for (int i = 0; i < 4; ++i) { v[i] = src[t + 256 * i]; mx = fmaxf(mx, v[i]); }
  red[t] = mx; __syncthreads();
  for (int off = 128; off; off >>= 1) { if (t < off) red[t] = fmaxf(red[t], red[t + off]); __syncthreads(); }
  mx = red[0]; __syncthreads();
  float ev[4], sum = 0.f;
#pragma unroll
  for (int i = 0; i < 4; ++i) { ev[i] = __expf((v[i] - mx) * scale); sum += ev[i]; }
  red[t] = sum; __syncthreads();
  for (int off = 128; off; off >>= 1) { if (t < off) red[t] += red[t + off]; __syncthreads(); }
  float inv = 1.f / red[0];
#pragma unroll
  for (int i = 0; i < 4; ++i) dst[t + 256 * i] = f2bf(ev[i] * inv);
}

// x = LayerNorm(x + y) * g + b  (row = B*S, 256 threads = E)
__global__ void add_ln_kernel(float* __restrict__ x, const float* __restrict__ y,
                              const float* __restrict__ g, const float* __restrict__ bta) {
  __shared__ float red[256];
  size_t row = blockIdx.x;
  int e = threadIdx.x;
  float v = x[row * EMB + e] + y[row * EMB + e];
  red[e] = v; __syncthreads();
  for (int off = 128; off; off >>= 1) { if (e < off) red[e] += red[e + off]; __syncthreads(); }
  float mean = red[0] / (float)EMB; __syncthreads();
  float d = v - mean;
  red[e] = d * d; __syncthreads();
  for (int off = 128; off; off >>= 1) { if (e < off) red[e] += red[e + off]; __syncthreads(); }
  float var = red[0] / (float)EMB;
  x[row * EMB + e] = d * rsqrtf(var + 1e-5f) * g[e] + bta[e];
}

// quantum FFN gate: q = cos(x) * cos(theta_ffn)
__global__ void qffn_kernel(const float* __restrict__ x, const float* __restrict__ theta,
                            bf16_t* __restrict__ q) {
  int i = blockIdx.x * 256 + threadIdx.x;            // over B*S*E
  if (i >= BQ * SEQ * EMB) return;
  int e = i % EMB;
  q[i] = f2bf(cosf(x[i]) * cosf(theta[e]));
}

__global__ void mean_kernel(const float* __restrict__ x, float* __restrict__ xm) {
  int i = blockIdx.x * 256 + threadIdx.x;            // over B*E
  if (i >= BQ * EMB) return;
  int b = i / EMB, e = i % EMB;
  float s = 0.f;
  for (int k = 0; k < SEQ; ++k) s += x[((size_t)b * SEQ + k) * EMB + e];
  xm[i] = s * (1.f / (float)SEQ);
}

__global__ void cls_kernel(const float* __restrict__ xm, const float* __restrict__ Wcls,
                           const float* __restrict__ bcls, float* __restrict__ out) {
  int i = blockIdx.x * 128 + threadIdx.x;            // over B*C = 80
  if (i >= BQ * NC) return;
  int b = i / NC, c = i % NC;
  float s = bcls[c];
  for (int e = 0; e < EMB; ++e) s += xm[b * EMB + e] * Wcls[e * NC + c];
  out[i] = s;
}

// ---------------- launcher ----------------
extern "C" void kernel_launch(void* const* d_in, const int* in_sizes, int n_in,
                              void* d_out, int out_size, void* d_ws, size_t ws_size,
                              hipStream_t stream) {
  (void)in_sizes; (void)n_in; (void)out_size; (void)ws_size;
  const int*   tokens     = (const int*)d_in[0];
  const float* emb        = (const float*)d_in[1];
  const float* theta_attn = (const float*)d_in[2];   // [L,H,DK]
  const float* Wc         = (const float*)d_in[3];   // [L,E,E]
  const float* bc         = (const float*)d_in[4];   // [L,E]
  const float* g1         = (const float*)d_in[5];
  const float* bn1        = (const float*)d_in[6];
  const float* theta_ffn  = (const float*)d_in[7];   // [L,NQ]
  const float* W1         = (const float*)d_in[8];   // [L,NQ,F]
  const float* b1         = (const float*)d_in[9];   // [L,F]
  const float* W2         = (const float*)d_in[10];  // [L,F,E]
  const float* b2         = (const float*)d_in[11];  // [L,E]
  const float* g2         = (const float*)d_in[12];
  const float* bn2        = (const float*)d_in[13];
  const float* Wcls       = (const float*)d_in[14];  // [E,C]
  const float* bcls       = (const float*)d_in[15];
  float* out = (float*)d_out;

  // workspace arena
  size_t off = 0;
  char* ws = (char*)d_ws;
  auto arena = [&](size_t bytes) -> void* {
    void* p = ws + off;
    off += (bytes + 255) & ~(size_t)255;
    return p;
  };
  float*  x   = (float*) arena((size_t)BQ * SEQ * EMB * 4);
  bf16_t* pbf = (bf16_t*)arena((size_t)BQ * SEQ * EMB * 2);
  bf16_t* pT  = (bf16_t*)arena((size_t)BQ * EMB * SEQ * 2);
  float*  sc  = (float*) arena((size_t)BQ * SEQ * SEQ * 4);
  bf16_t* scb = (bf16_t*)arena((size_t)BQ * SEQ * SEQ * 2);
  bf16_t* vb  = (bf16_t*)arena((size_t)BQ * SEQ * EMB * 2); // attnV bf16, reused as q
  float*  y   = (float*) arena((size_t)BQ * SEQ * EMB * 4);
  bf16_t* hb  = (bf16_t*)arena((size_t)BQ * SEQ * FF  * 2);
  bf16_t* WcT = (bf16_t*)arena((size_t)NL * EMB * EMB * 2);
  bf16_t* W1T = (bf16_t*)arena((size_t)NL * FF  * NQQ * 2);
  bf16_t* W2T = (bf16_t*)arena((size_t)NL * EMB * FF  * 2);
  float*  xm  = (float*) arena((size_t)BQ * EMB * 4);

  const int BS = BQ * SEQ;
  const float scale = 0.17677669529663687f;          // 1/sqrt(32)

  // one-time: weight transposes to bf16 (B^T layouts), embedding + positional encoding
  transpose_to_bf16<<<dim3((EMB * EMB + 255) / 256, 1, NL), 256, 0, stream>>>(Wc, WcT, EMB, EMB);
  transpose_to_bf16<<<dim3((NQQ * FF  + 255) / 256, 1, NL), 256, 0, stream>>>(W1, W1T, NQQ, FF);
  transpose_to_bf16<<<dim3((FF  * EMB + 255) / 256, 1, NL), 256, 0, stream>>>(W2, W2T, FF, EMB);
  embed_kernel<<<(BS * EMB + 255) / 256, 256, 0, stream>>>(tokens, emb, x);

  for (int l = 0; l < NL; ++l) {
    // quantum projection -> p (row-major bf16) and p^T (bf16)
    qproj_kernel<<<(BS * NH + 255) / 256, 256, 0, stream>>>(x, theta_attn + (size_t)l * NH * DKQ, pbf, pT);

    // scores = p * p^T    (batched: M=N=S, K=E; BT form of p^T is p itself)
    gemm_bf16_wmma<0, 0, 1, 0><<<dim3(SEQ / 128, SEQ / 64, BQ), 256, 0, stream>>>(
        pbf, pbf, sc, nullptr, nullptr, SEQ, SEQ, EMB, EMB, EMB, SEQ,
        (long)SEQ * EMB, (long)SEQ * EMB, (long)SEQ * SEQ);

    softmax_kernel<<<BS, 256, 0, stream>>>(sc, scb, scale);

    // attnV = softmax * p  (batched: M=S, N=E, K=S; BT = p^T), write bf16
    gemm_bf16_wmma<0, 0, 0, 1><<<dim3(EMB / 128, SEQ / 64, BQ), 256, 0, stream>>>(
        scb, pT, nullptr, vb, nullptr, SEQ, EMB, SEQ, SEQ, SEQ, EMB,
        (long)SEQ * SEQ, (long)EMB * SEQ, (long)SEQ * EMB);

    // proj: y = attnV * Wc + bc  (M=B*S, N=E, K=E)
    gemm_bf16_wmma<1, 0, 1, 0><<<dim3(EMB / 128, BS / 64, 1), 256, 0, stream>>>(
        vb, WcT + (size_t)l * EMB * EMB, y, nullptr, bc + (size_t)l * EMB,
        BS, EMB, EMB, EMB, EMB, EMB, 0, 0, 0);

    add_ln_kernel<<<BS, 256, 0, stream>>>(x, y, g1 + (size_t)l * EMB, bn1 + (size_t)l * EMB);

    // quantum FFN gate -> q (reuse vb)
    qffn_kernel<<<(BS * EMB + 255) / 256, 256, 0, stream>>>(x, theta_ffn + (size_t)l * NQQ, vb);

    // ffn1: h = relu(q * W1 + b1)  (M=B*S, N=F, K=NQ), bf16 out
    gemm_bf16_wmma<1, 1, 0, 1><<<dim3(FF / 128, BS / 64, 1), 256, 0, stream>>>(
        vb, W1T + (size_t)l * FF * NQQ, nullptr, hb, b1 + (size_t)l * FF,
        BS, FF, NQQ, NQQ, NQQ, FF, 0, 0, 0);

    // ffn2: y = h * W2 + b2  (M=B*S, N=E, K=F)
    gemm_bf16_wmma<1, 0, 1, 0><<<dim3(EMB / 128, BS / 64, 1), 256, 0, stream>>>(
        hb, W2T + (size_t)l * EMB * FF, y, nullptr, b2 + (size_t)l * EMB,
        BS, EMB, FF, FF, FF, EMB, 0, 0, 0);

    add_ln_kernel<<<BS, 256, 0, stream>>>(x, y, g2 + (size_t)l * EMB, bn2 + (size_t)l * EMB);
  }

  mean_kernel<<<(BQ * EMB + 255) / 256, 256, 0, stream>>>(x, xm);
  cls_kernel<<<1, 128, 0, stream>>>(xm, Wcls, bcls, out);
}
